// cheb_GCN_67740224192528
// MI455X (gfx1250) — matrix operand order repaired and checked by reading, not compiled, not run on hardware
//
#include <hip/hip_runtime.h>
#include <hip/hip_bf16.h>

typedef __attribute__((ext_vector_type(16))) _Float16 v16h;
typedef __attribute__((ext_vector_type(8)))  _Float16 v8h;
typedef __attribute__((ext_vector_type(8)))  float    v8f;

static inline int cdiv(long long a, long long b) { return (int)((a + b - 1) / b); }

// ---------------- small utility kernels ----------------

__global__ void zero_f32_kernel(float* p, int n) {
    int i = blockIdx.x * blockDim.x + threadIdx.x;
    if (i < n) p[i] = 0.f;
}

__global__ void deg_accum_kernel(const int* __restrict__ er, float* __restrict__ deg, int E) {
    int i = blockIdx.x * blockDim.x + threadIdx.x;
    if (i < E) atomicAdd(&deg[er[i]], 1.0f);
}

__global__ void dinv_kernel(float* d, int n) {
    int i = blockIdx.x * blockDim.x + threadIdx.x;
    if (i < n) {
        float v = d[i];
        d[i] = (v > 0.f) ? rsqrtf(fmaxf(v, 1.f)) : 0.f;
    }
}

__global__ void f32_to_f16_kernel(const float* __restrict__ s, _Float16* __restrict__ d, int n) {
    int i = blockIdx.x * blockDim.x + threadIdx.x;
    if (i < n) d[i] = (_Float16)s[i];
}

// Pack f32 weight matrix [Kreal(,pad to Kpad), Fout] row-major into per-wave WMMA
// B-fragment order: dst[((ct*numKb + kb)*32 + lane)*16 + p] = B[kb*32 + (lane>>4)*16 + p][ct*16 + (lane&15)]
// (this code's contiguous k order, matching A: lane (n,hi) holds K = hi*16 + 0..15, linear).
__global__ void pack_b_kernel(const float* __restrict__ W, _Float16* __restrict__ dst,
                              int Kreal, int Kpad, int Fout) {
    int numKb = Kpad / 32;
    int total = (Fout / 16) * numKb * 512;
    int i = blockIdx.x * blockDim.x + threadIdx.x;
    if (i >= total) return;
    int p    = i & 15;
    int lane = (i >> 4) & 31;
    int kb   = (i >> 9) % numKb;
    int ct   = i / (numKb * 512);
    int k    = kb * 32 + (lane >> 4) * 16 + p;
    int col  = ct * 16 + (lane & 15);
    float v  = (k < Kreal) ? W[(size_t)k * Fout + col] : 0.f;
    dst[i] = (_Float16)v;
}

// Fill Tcat slice 0 (= x itself) and zero the K padding columns
__global__ void cheb_slice0_kernel(const float* __restrict__ xin, _Float16* __restrict__ tcat,
                                   int Nn, int Fin, int Kpad, int B) {
    int i = blockIdx.x * blockDim.x + threadIdx.x;
    int total = B * Nn * Kpad;
    if (i >= total) return;
    int col = i % Kpad;
    int n   = (i / Kpad) % Nn;
    int b   = i / (Kpad * Nn);
    if (col < Fin) {
        tcat[i] = (_Float16)xin[((size_t)b * Nn + n) * Fin + col];
    } else if (col >= 6 * Fin) {
        tcat[i] = (_Float16)0.f;   // K padding
    }
    // cols [Fin, 6*Fin) are written by the spmv slices afterwards
}

// Gather-form Chebyshev SpMV: fixed degree 6 (er = repeat(arange(N),6)).
//   s = L(tin)[b,n,f];  out = (tpp ? 2*s - tpp : s);  also emit f16 Tcat slice.
__global__ void cheb_spmv_kernel(const float* __restrict__ tin, const float* __restrict__ tpp,
                                 const int* __restrict__ ec, const float* __restrict__ dinv,
                                 float* __restrict__ tout, _Float16* __restrict__ tcat,
                                 int Nn, int Fin, int Kpad, int slice, int B) {
    int i = blockIdx.x * blockDim.x + threadIdx.x;
    int total = B * Nn * Fin;
    if (i >= total) return;
    int f = i % Fin;
    int n = (i / Fin) % Nn;
    int b = i / (Fin * Nn);
    const int* ep = ec + n * 6;
    float s = 0.f;
#pragma unroll
    for (int j = 0; j < 6; ++j) {
        int c = ep[j];
        s += dinv[c] * tin[((size_t)b * Nn + c) * Fin + f];
    }
    s *= -dinv[n];                       // norm = -(dinv[er]*dinv[ec])
    if (tpp) s = 2.f * s - tpp[i];       // Tx_k = 2 L(Tx_{k-1}) - Tx_{k-2}
    tout[i] = s;
    tcat[((size_t)b * Nn + n) * Kpad + slice * Fin + f] = (_Float16)s;
}

// Fixed-nnz (3) pooling gather: pooled[b,m,f] = sum_t pv[3m+t] * xin[b, pc[3m+t], f]
__global__ void pool_kernel(const float* __restrict__ xin, const int* __restrict__ pc,
                            const float* __restrict__ pv, float* __restrict__ out,
                            int Nn, int Mm, int F, int B) {
    int i = blockIdx.x * blockDim.x + threadIdx.x;
    int total = B * Mm * F;
    if (i >= total) return;
    int f = i % F;
    int m = (i / F) % Mm;
    int b = i / (F * Mm);
    float s = 0.f;
#pragma unroll
    for (int t = 0; t < 3; ++t) {
        int e = m * 3 + t;
        s += pv[e] * xin[((size_t)b * Nn + pc[e]) * F + f];
    }
    out[i] = s;
}

// ---------------- WMMA GEMM (f16 A/B, f32 acc) with fused bias + optional ReLU ----
// C[rows, Ncols] = relu(A[rows, K] * Bpack + bias).  rows%16==0, Ncols%16==0, K%32==0.
// A is row-major [rows, K] f16; B is pre-packed in fragment order by pack_b_kernel.
// One 16x16 output tile per wave32; C/D layout per CDNA5 ISA 7.12.2; A/B in contiguous k order.
// A loads: 2x b128 per lane per k-block (contiguous runs); B loads: 2x b128, coalesced.
__global__ void wmma_gemm_bias_relu(const _Float16* __restrict__ A, const _Float16* __restrict__ Bp,
                                    const float* __restrict__ bias, float* __restrict__ C,
                                    int rows, int K, int Ncols, int doRelu) {
    const int lane  = threadIdx.x & 31;
    const int wave  = threadIdx.x >> 5;
    const int tile  = blockIdx.x * (blockDim.x >> 5) + wave;   // wave-uniform
    const int rowTiles = rows >> 4;
    const int colTiles = Ncols >> 4;
    if (tile >= rowTiles * colTiles) return;                   // uniform exit: EXEC stays all-1s
    const int rt = tile / colTiles;
    const int ct = tile % colTiles;
    const int numKb = K >> 5;

    const int m  = lane & 15;
    const int hi = lane >> 4;          // which 16-lane half
    const _Float16* Arow  = A + (size_t)(rt * 16 + m) * K + hi * 16;
    const _Float16* Blane = Bp + ((size_t)ct * numKb * 32 + lane) * 16;

    v8f acc = {};
    for (int kb = 0; kb < numKb; ++kb) {
        const int k0 = kb << 5;
        __builtin_prefetch(Arow + k0 + 32, 0, 1);  // next k-block (global_prefetch_b8)
        // A 16x32 f16 fragment: lane holds K = hi*16+0..15 (two b128 runs)
        v8h a0 = *(const v8h*)(Arow + k0);
        v8h a1 = *(const v8h*)(Arow + k0 + 8);
        v16h a = __builtin_shufflevector(a0, a1, 0, 1, 2, 3, 4, 5, 6, 7,
                                         8, 9, 10, 11, 12, 13, 14, 15);
        // B 32x16 f16 fragment: packed contiguous 16 halves per lane (two b128 runs)
        const v8h* bp = (const v8h*)(Blane + (size_t)kb * 512);
        v16h b = __builtin_shufflevector(bp[0], bp[1], 0, 1, 2, 3, 4, 5, 6, 7,
                                         8, 9, 10, 11, 12, 13, 14, 15);
        acc = __builtin_amdgcn_wmma_f32_16x16x32_f16(false, a, false, b, (short)0, acc,
                                                     false, false);
    }
    // C/D layout: VGPR j -> row j (lanes 0-15) / row j+8 (lanes 16-31), col = lane&15
    const int col = ct * 16 + (lane & 15);
    const float bv = bias[col];
#pragma unroll
    for (int j = 0; j < 8; ++j) {
        int row = rt * 16 + j + hi * 8;
        float v = acc[j] + bv;
        if (doRelu) v = fmaxf(v, 0.f);
        C[(size_t)row * Ncols + col] = v;
    }
}

// classifier: out[b,j] = h[b,:] . clsW[:,j] + clsb[j]   (32x10, not WMMA-shaped)
__global__ void cls_kernel(const float* __restrict__ h, const float* __restrict__ W,
                           const float* __restrict__ b, float* __restrict__ out) {
    int i = blockIdx.x * blockDim.x + threadIdx.x;
    if (i >= 32 * 10) return;
    int bi = i / 10, j = i % 10;
    float s = b[j];
#pragma unroll 8
    for (int f = 0; f < 128; ++f) s += h[bi * 128 + f] * W[f * 10 + j];
    out[i] = s;
}

// ---------------- host orchestration ----------------

extern "C" void kernel_launch(void* const* d_in, const int* in_sizes, int n_in,
                              void* d_out, int out_size, void* d_ws, size_t ws_size,
                              hipStream_t stream) {
    (void)in_sizes; (void)n_in; (void)out_size; (void)ws_size;
    const int B = 32;
    const int Nn[4]   = {16384, 4096, 1024, 256};
    const int Mm[4]   = {4096, 1024, 256, 64};
    const int Fin[4]  = {3, 32, 32, 32};
    const int Fout[4] = {32, 32, 32, 64};
    const int Kpad[4] = {32, 192, 192, 192};   // pad 6*Fin up to multiple of 32

    // input pointers (setup_inputs dict order)
    const float* x0 = (const float*)d_in[0];
    const int* er[4] = {(const int*)d_in[1], (const int*)d_in[3], (const int*)d_in[5], (const int*)d_in[7]};
    const int* ec[4] = {(const int*)d_in[2], (const int*)d_in[4], (const int*)d_in[6], (const int*)d_in[8]};
    const int*   pc[4] = {(const int*)d_in[10], (const int*)d_in[13], (const int*)d_in[16], (const int*)d_in[19]};
    const float* pv[4] = {(const float*)d_in[11], (const float*)d_in[14], (const float*)d_in[17], (const float*)d_in[20]};
    const float* Wl[4] = {(const float*)d_in[21], (const float*)d_in[23], (const float*)d_in[25], (const float*)d_in[27]};
    const float* bl[4] = {(const float*)d_in[22], (const float*)d_in[24], (const float*)d_in[26], (const float*)d_in[28]};
    const float* encW = (const float*)d_in[29];
    const float* encB = (const float*)d_in[30];
    const float* clsW = (const float*)d_in[31];
    const float* clsB = (const float*)d_in[32];
    float* out = (float*)d_out;

    // workspace carve (256B aligned)
    char* wp = (char*)d_ws;
    auto carve = [&](size_t bytes) -> void* {
        void* p = (void*)wp;
        wp += (bytes + 255) & ~(size_t)255;
        return p;
    };
    float*     dinv   = (float*)carve((size_t)16384 * 4);
    _Float16*  tcat   = (_Float16*)carve((size_t)25165824 * 2);  // max B*N*Kpad (layer 1)
    float*     txA    = (float*)carve((size_t)4194304 * 4);      // max B*N*Fin
    float*     txB    = (float*)carve((size_t)4194304 * 4);
    float*     xout   = (float*)carve((size_t)16777216 * 4);     // max B*N*Fout (layer 0)
    float*     poolA  = (float*)carve((size_t)4194304 * 4);      // max B*M*Fout
    float*     poolB  = (float*)carve((size_t)4194304 * 4);
    _Float16*  wcatP  = (_Float16*)carve((size_t)192 * 64 * 2);       // packed layer weights
    _Float16*  encWp  = (_Float16*)carve((size_t)4096 * 128 * 2);     // packed encoder weights
    _Float16*  xflatH = (_Float16*)carve((size_t)32 * 4096 * 2);
    float*     hbuf   = (float*)carve((size_t)32 * 128 * 4);

    const int T = 256;
    const float* xin = x0;
    float* poolPing[2] = {poolA, poolB};

    for (int L = 0; L < 4; ++L) {
        const int N = Nn[L], M = Mm[L], fi = Fin[L], fo = Fout[L], kp = Kpad[L];
        const int E = N * 6;

        // dinv = D^{-1/2}
        zero_f32_kernel<<<cdiv(N, T), T, 0, stream>>>(dinv, N);
        deg_accum_kernel<<<cdiv(E, T), T, 0, stream>>>(er[L], dinv, E);
        dinv_kernel<<<cdiv(N, T), T, 0, stream>>>(dinv, N);

        // weights -> packed f16 B fragments [colTiles][numKb][lane][16]
        pack_b_kernel<<<cdiv((fo / 16) * (kp / 32) * 512, T), T, 0, stream>>>(Wl[L], wcatP,
                                                                              6 * fi, kp, fo);

        // Tcat slice 0 (+ zero padding columns)
        cheb_slice0_kernel<<<cdiv((long long)B * N * kp, T), T, 0, stream>>>(xin, tcat, N, fi, kp, B);

        // Chebyshev recurrence: Tx1 = L(x); Tx_k = 2 L(Tx_{k-1}) - Tx_{k-2}
        const int nel = B * N * fi;
        cheb_spmv_kernel<<<cdiv(nel, T), T, 0, stream>>>(xin, nullptr, ec[L], dinv, txA, tcat,
                                                         N, fi, kp, 1, B);
        const float* tprev = txA;       // Tx1
        const float* tpp   = xin;       // Tx0
        float* touts[4] = {txB, txA, txB, txA};
        for (int k = 2; k <= 5; ++k) {
            float* to = touts[k - 2];
            cheb_spmv_kernel<<<cdiv(nel, T), T, 0, stream>>>(tprev, tpp, ec[L], dinv, to, tcat,
                                                             N, fi, kp, k, B);
            tpp = tprev;
            tprev = to;
        }

        // GEMM: [B*N, Kpad] x [Kpad, Fout] -> xout, fused bias+ReLU (WMMA)
        {
            int rows = B * N;
            int waves = (rows / 16) * (fo / 16);
            wmma_gemm_bias_relu<<<cdiv(waves, T / 32), T, 0, stream>>>(tcat, wcatP, bl[L], xout,
                                                                      rows, kp, fo, 1);
        }

        // Pool -> next level input
        float* pooled = poolPing[L & 1];
        pool_kernel<<<cdiv(B * M * fo, T), T, 0, stream>>>(xout, pc[L], pv[L], pooled, N, M, fo, B);
        xin = pooled;
    }

    // encoder: flatten [32, 64*64] -> f16 A, packed f16 B, WMMA GEMM to 128, ReLU
    f32_to_f16_kernel<<<cdiv(32 * 4096, T), T, 0, stream>>>(xin, xflatH, 32 * 4096);
    pack_b_kernel<<<cdiv((128 / 16) * (4096 / 32) * 512, T), T, 0, stream>>>(encW, encWp,
                                                                             4096, 4096, 128);
    {
        int waves = (32 / 16) * (128 / 16);   // 16 waves
        wmma_gemm_bias_relu<<<cdiv(waves, T / 32), T, 0, stream>>>(xflatH, encWp, encB, hbuf,
                                                                   32, 4096, 128, 1);
    }
    // classifier 128 -> 10
    cls_kernel<<<cdiv(320, T), T, 0, stream>>>(hbuf, clsW, clsB, out);
}
